// FlowActionHeadPACE_50938312131045
// MI455X (gfx1250) — matrix operand order repaired and checked by reading, not compiled, not run on hardware
//
#include <hip/hip_runtime.h>

// ---------------- problem constants ----------------
#define Bsz        4096
#define FUSION     1024
#define SKILL      128
#define CONT       64
#define LATENT     512
#define NEXP       8
#define HID        1024
#define COND_IN    1216          // FUSION + SKILL + CONT (38*32)
#define XPAD       1568          // pad EXPERT_IN=1537 up to 49*32
#define FLOW_STEPS 8
#define OUT_D      512           // TA*DA

typedef __bf16 bf16;
typedef __attribute__((ext_vector_type(16))) __bf16 v16bf;
typedef __attribute__((ext_vector_type(8)))  __bf16 v8bf;
typedef __attribute__((ext_vector_type(8)))  float  v8f;

static __device__ __forceinline__ float silu_f(float x) {
  return x / (1.0f + __expf(-x));
}

// ---------------------------------------------------------------------------
// NT 16x16 output tiles sharing the same 16 A-rows:
//   D[t] = A(16 x 32*ksteps) * B[t](32*ksteps x 16)
// A: row-major bf16 (two contiguous 16B loads/lane, contiguous k = 16*half + s, same as B).
// B: TRANSPOSED weights Bt[n][k]; one contiguous 32B load/lane/tile.
// Software-pipelined one k-step ahead: loads for kk+1 are issued before the
// wmma chain of kk, forcing two fragment generations to stay live so the
// backend can overlap L2 return latency with matrix ops (partial
// s_wait_loadcnt instead of 0).  unroll 2 resolves the rotation by renaming.
// ---------------------------------------------------------------------------
template <int NT>
static __device__ __forceinline__ void
wmma_multitile(const bf16* __restrict__ A, int lda,
               const bf16* __restrict__ Bt, int ldb,
               int ksteps, int lane, v8f* acc) {
  const int r    = lane & 15;
  const int half = lane >> 4;
  const bf16* ap = A  + (size_t)r * lda + half * 16;
  const bf16* bp = Bt + (size_t)r * ldb + half * 16;
  const size_t bstride = (size_t)16 * ldb;     // next 16-col tile

  // prologue: fragments for k-step 0
  v8bf lo = *(const v8bf*)(ap);
  v8bf hi = *(const v8bf*)(ap + 8);
  v16bf b[NT];
#pragma unroll
  for (int t = 0; t < NT; ++t)
    b[t] = *(const v16bf*)(bp + (size_t)t * bstride);

#pragma unroll 2
  for (int kk = 0; kk < ksteps - 1; ++kk) {
    ap += 32;
    bp += 32;
    // issue next k-step's loads first (independent of the wmmas below)
    v8bf lo_n = *(const v8bf*)(ap);
    v8bf hi_n = *(const v8bf*)(ap + 8);
    v16bf b_n[NT];
#pragma unroll
    for (int t = 0; t < NT; ++t)
      b_n[t] = *(const v16bf*)(bp + (size_t)t * bstride);
    // consume current fragments
    v16bf a;
#pragma unroll
    for (int i = 0; i < 8; ++i) { a[i] = lo[i]; a[i + 8] = hi[i]; }
#pragma unroll
    for (int t = 0; t < NT; ++t)
      acc[t] = __builtin_amdgcn_wmma_f32_16x16x32_bf16(
          false, a, false, b[t], (short)0, acc[t], false, false);
    // rotate
    lo = lo_n;
    hi = hi_n;
#pragma unroll
    for (int t = 0; t < NT; ++t) b[t] = b_n[t];
  }
  // epilogue: last k-step
  {
    v16bf a;
#pragma unroll
    for (int i = 0; i < 8; ++i) { a[i] = lo[i]; a[i + 8] = hi[i]; }
#pragma unroll
    for (int t = 0; t < NT; ++t)
      acc[t] = __builtin_amdgcn_wmma_f32_16x16x32_bf16(
          false, a, false, b[t], (short)0, acc[t], false, false);
  }
}

// ---------------- prep kernels ----------------

// concat [fused_obs | phase | skill] -> bf16 xc[B][COND_IN]
__global__ void build_xc(const float* __restrict__ fo, const float* __restrict__ pe,
                         const float* __restrict__ sl, bf16* __restrict__ xc) {
  int idx = blockIdx.x * blockDim.x + threadIdx.x;
  if (idx >= Bsz * COND_IN) return;
  int b = idx / COND_IN, c = idx - b * COND_IN;
  float v;
  if (c < FUSION)              v = fo[b * FUSION + c];
  else if (c < FUSION + SKILL) v = pe[b * SKILL + (c - FUSION)];
  else                         v = sl[b * CONT + (c - FUSION - SKILL)];
  xc[idx] = (bf16)v;
}

// Coalesced LDS-tiled transpose: W[e][Kdim][N] f32 -> Wt[e][N][KPAD] bf16,
// zero-padded in K.  grid = (KPAD/32, N/32, E), 256 threads.
__global__ void __launch_bounds__(256)
transpose_tile_bf16(const float* __restrict__ W, bf16* __restrict__ Wt,
                    int Kdim, int N, int KPAD) {
  __shared__ float tile[32][33];
  const int e  = blockIdx.z;
  const int k0 = blockIdx.x * 32;
  const int n0 = blockIdx.y * 32;
  const int tx = threadIdx.x & 31;
  const int ty = threadIdx.x >> 5;           // 0..7
  const float* Wsrc = W + (size_t)e * Kdim * N;
#pragma unroll
  for (int i = 0; i < 4; ++i) {
    int k = k0 + ty + i * 8;
    float v = (k < Kdim) ? Wsrc[(size_t)k * N + (n0 + tx)] : 0.0f;  // coalesced
    tile[ty + i * 8][tx] = v;
  }
  __syncthreads();
  bf16* Wdst = Wt + (size_t)e * N * KPAD;
#pragma unroll
  for (int i = 0; i < 4; ++i) {
    int n = n0 + ty + i * 8;
    Wdst[(size_t)n * KPAD + (k0 + tx)] = (bf16)tile[tx][ty + i * 8];  // coalesced
  }
}

// x[b][0:512] = u0 (bf16), u(f32)=u0; tau col (1536) = 0; pad cols = 0
__global__ void init_x(const float* __restrict__ u0, bf16* __restrict__ x,
                       float* __restrict__ u) {
  int idx = blockIdx.x * blockDim.x + threadIdx.x;
  if (idx >= Bsz * XPAD) return;
  int b = idx / XPAD, c = idx - b * XPAD;
  if (c < LATENT) {
    float v = u0[b * LATENT + c];
    x[idx] = (bf16)v;
    u[b * LATENT + c] = v;
  } else if (c >= 1536) {
    x[idx] = (bf16)0.0f;      // tau for step 0 is 0; pad is 0
  }
}

// ---------------- conditioner GEMM: cond = xc @ Wc + bc -> x[:,512:1536] ----
__global__ void __launch_bounds__(256)
cond_kernel(const bf16* __restrict__ xc, const bf16* __restrict__ Wct,
            const float* __restrict__ bc, bf16* __restrict__ x) {
  const int lane = threadIdx.x & 31;
  const int wave = threadIdx.x >> 5;
  const int row0 = blockIdx.x * 16;
  const int r = lane & 15, half = lane >> 4;
  v8f z = {};
  v8f acc[8];
#pragma unroll
  for (int t = 0; t < 8; ++t) acc[t] = z;
  wmma_multitile<8>(xc + (size_t)row0 * COND_IN, COND_IN,
                    Wct + (size_t)(wave * 128) * COND_IN, COND_IN,
                    COND_IN / 32, lane, acc);
#pragma unroll
  for (int t = 0; t < 8; ++t)
#pragma unroll
    for (int v = 0; v < 8; ++v) {
      int row = half * 8 + v, col = wave * 128 + t * 16 + r;
      x[(size_t)(row0 + row) * XPAD + 512 + col] = (bf16)(acc[t][v] + bc[col]);
    }
}

// ---------------- fused flow step: all 8 experts + gate + Euler update ------
// Block owns 16 rows; loops experts, h1/h2 staged in LDS, gated sum of v_e
// kept in registers, then u += dt*v and the bf16 u-slice of x is rewritten.
__global__ void __launch_bounds__(256)
step_kernel(bf16* x,
            const bf16* __restrict__ W1t, const float* __restrict__ b1,
            const bf16* __restrict__ W2t, const float* __restrict__ b2,
            const bf16* __restrict__ W3t, const float* __restrict__ b3,
            const float* __restrict__ p_hat, float* __restrict__ u,
            float dt, float tau_next) {
  __shared__ bf16 h1[16 * HID];   // 32 KB
  __shared__ bf16 h2[16 * HID];   // 32 KB
  const int lane = threadIdx.x & 31;
  const int wave = threadIdx.x >> 5;
  const int row0 = blockIdx.x * 16;
  const int r = lane & 15, half = lane >> 4;
  const v8f z = {};

  v8f vacc[4];
#pragma unroll
  for (int t = 0; t < 4; ++t) vacc[t] = z;

  const bf16* Abase = x + (size_t)row0 * XPAD;

  for (int k = 0; k < NEXP; ++k) {
    // ---- GEMM1: h1 = silu(x @ W1 + b1), K = XPAD (zero-padded) ----
    {
      v8f acc[8];
#pragma unroll
      for (int t = 0; t < 8; ++t) acc[t] = z;
      wmma_multitile<8>(Abase, XPAD,
                        W1t + ((size_t)k * HID + wave * 128) * XPAD, XPAD,
                        XPAD / 32, lane, acc);
#pragma unroll
      for (int t = 0; t < 8; ++t)
#pragma unroll
        for (int v = 0; v < 8; ++v) {
          int row = half * 8 + v, col = wave * 128 + t * 16 + r;
          h1[row * HID + col] = (bf16)silu_f(acc[t][v] + b1[k * HID + col]);
        }
    }
    __syncthreads();
    // ---- GEMM2: h2 = silu(h1 @ W2 + b2) ----
    {
      v8f acc[8];
#pragma unroll
      for (int t = 0; t < 8; ++t) acc[t] = z;
      wmma_multitile<8>(h1, HID,
                        W2t + ((size_t)k * HID + wave * 128) * HID, HID,
                        HID / 32, lane, acc);
#pragma unroll
      for (int t = 0; t < 8; ++t)
#pragma unroll
        for (int v = 0; v < 8; ++v) {
          int row = half * 8 + v, col = wave * 128 + t * 16 + r;
          h2[row * HID + col] = (bf16)silu_f(acc[t][v] + b2[k * HID + col]);
        }
    }
    __syncthreads();
    // ---- GEMM3: vacc += gate_k * (h2 @ W3 + b3) ----
    {
      v8f acc[4];
#pragma unroll
      for (int t = 0; t < 4; ++t) acc[t] = z;
      wmma_multitile<4>(h2, HID,
                        W3t + ((size_t)k * LATENT + wave * 64) * HID, HID,
                        HID / 32, lane, acc);
#pragma unroll
      for (int t = 0; t < 4; ++t)
#pragma unroll
        for (int v = 0; v < 8; ++v) {
          int row = half * 8 + v, col = wave * 64 + t * 16 + r;
          float g = p_hat[(row0 + row) * NEXP + k];
          vacc[t][v] += g * (acc[t][v] + b3[k * LATENT + col]);
        }
    }
    // next expert's post-GEMM1 barrier orders these h2 reads vs. next h2 writes
  }

  __syncthreads();   // all waves done reading x rows before in-place rewrite

  // ---- Euler update: u += dt*v ; refresh bf16 u-slice of x + tau column ----
#pragma unroll
  for (int t = 0; t < 4; ++t)
#pragma unroll
    for (int v = 0; v < 8; ++v) {
      int row = half * 8 + v, col = wave * 64 + t * 16 + r;
      int brow = row0 + row;
      size_t idx = (size_t)brow * LATENT + col;
      float un = u[idx] + dt * vacc[t][v];
      u[idx] = un;
      x[(size_t)brow * XPAD + col] = (bf16)un;
    }
  if (wave == 0 && r == 0) {
#pragma unroll
    for (int v = 0; v < 8; ++v) {
      int brow = row0 + half * 8 + v;
      x[(size_t)brow * XPAD + 1536] = (bf16)tau_next;
    }
  }
}

// ---------------- decoder GEMM: out = u @ Wd + bd ; also emit final u -------
__global__ void __launch_bounds__(256)
dec_kernel(const bf16* __restrict__ x, const bf16* __restrict__ Wdt,
           const float* __restrict__ bd, const float* __restrict__ u,
           float* __restrict__ out, float* __restrict__ u_out) {
  const int lane = threadIdx.x & 31;
  const int wave = threadIdx.x >> 5;
  const int row0 = blockIdx.x * 16;
  const int r = lane & 15, half = lane >> 4;
  v8f z = {};
  v8f acc[4];
#pragma unroll
  for (int t = 0; t < 4; ++t) acc[t] = z;
  wmma_multitile<4>(x + (size_t)row0 * XPAD, XPAD,
                    Wdt + (size_t)(wave * 64) * LATENT, LATENT,
                    LATENT / 32, lane, acc);
#pragma unroll
  for (int t = 0; t < 4; ++t)
#pragma unroll
    for (int v = 0; v < 8; ++v) {
      int row = half * 8 + v, col = wave * 64 + t * 16 + r;
      size_t idx = (size_t)(row0 + row) * OUT_D + col;
      out[idx] = acc[t][v] + bd[col];
      u_out[idx] = u[idx];           // same (B,512) shape as decoder output
    }
}

// ---------------- launch ----------------
extern "C" void kernel_launch(void* const* d_in, const int* in_sizes, int n_in,
                              void* d_out, int out_size, void* d_ws, size_t ws_size,
                              hipStream_t stream) {
  const float* fused = (const float*)d_in[0];
  const float* phase = (const float*)d_in[1];
  const float* skill = (const float*)d_in[2];
  const float* p_hat = (const float*)d_in[3];
  const float* u0    = (const float*)d_in[4];
  const float* Wc    = (const float*)d_in[5];
  const float* bc    = (const float*)d_in[6];
  const float* W1    = (const float*)d_in[7];
  const float* b1    = (const float*)d_in[8];
  const float* W2    = (const float*)d_in[9];
  const float* b2    = (const float*)d_in[10];
  const float* W3    = (const float*)d_in[11];
  const float* b3    = (const float*)d_in[12];
  const float* Wd    = (const float*)d_in[13];
  const float* bd    = (const float*)d_in[14];

  char* ws = (char*)d_ws;
  size_t off = 0;
  auto alloc = [&](size_t bytes) -> void* {
    void* p = ws + off;
    off = (off + bytes + 255) & ~(size_t)255;
    return p;
  };
  bf16*  xc  = (bf16*)alloc((size_t)Bsz * COND_IN * 2);
  bf16*  x   = (bf16*)alloc((size_t)Bsz * XPAD * 2);
  bf16*  Wct = (bf16*)alloc((size_t)FUSION * COND_IN * 2);
  bf16*  W1t = (bf16*)alloc((size_t)NEXP * HID * XPAD * 2);
  bf16*  W2t = (bf16*)alloc((size_t)NEXP * HID * HID * 2);
  bf16*  W3t = (bf16*)alloc((size_t)NEXP * LATENT * HID * 2);
  bf16*  Wdt = (bf16*)alloc((size_t)OUT_D * LATENT * 2);
  float* u   = (float*)alloc((size_t)Bsz * LATENT * 4);

  const int T = 256;
  build_xc<<<(Bsz * COND_IN + T - 1) / T, T, 0, stream>>>(fused, phase, skill, xc);

  transpose_tile_bf16<<<dim3(COND_IN / 32, FUSION / 32, 1),    T, 0, stream>>>(Wc, Wct, COND_IN, FUSION, COND_IN);
  transpose_tile_bf16<<<dim3(XPAD / 32,    HID / 32,    NEXP), T, 0, stream>>>(W1, W1t, 1537,    HID,    XPAD);
  transpose_tile_bf16<<<dim3(HID / 32,     HID / 32,    NEXP), T, 0, stream>>>(W2, W2t, HID,     HID,    HID);
  transpose_tile_bf16<<<dim3(HID / 32,     LATENT / 32, NEXP), T, 0, stream>>>(W3, W3t, HID,     LATENT, HID);
  transpose_tile_bf16<<<dim3(LATENT / 32,  OUT_D / 32,  1),    T, 0, stream>>>(Wd, Wdt, LATENT,  OUT_D,  LATENT);

  init_x<<<(Bsz * XPAD + T - 1) / T, T, 0, stream>>>(u0, x, u);
  cond_kernel<<<Bsz / 16, T, 0, stream>>>(xc, Wct, bc, x);

  const float dt = 1.0f / FLOW_STEPS;
  for (int s = 0; s < FLOW_STEPS; ++s) {
    step_kernel<<<Bsz / 16, T, 0, stream>>>(
        x, W1t, b1, W2t, b2, W3t, b3, p_hat, u, dt, dt * (float)(s + 1));
  }

  dec_kernel<<<Bsz / 16, T, 0, stream>>>(
      x, Wdt, bd, u, (float*)d_out, (float*)d_out + (size_t)Bsz * OUT_D);
}
